// StructureBlock_32710470926528
// MI455X (gfx1250) — matrix operand order repaired and checked by reading, not compiled, not run on hardware
//
#include <hip/hip_runtime.h>
#include <hip/hip_bf16.h>
#include <math.h>

typedef __attribute__((ext_vector_type(16))) _Float16 v16h;
typedef __attribute__((ext_vector_type(8)))  _Float16 v8h;
typedef __attribute__((ext_vector_type(8)))  float    v8f;

#define N_RES 768
#define CN 384
#define CZ 128
#define CH 16
#define NH 12
#define PQ 4
#define PV 8
#define CATD 2112   /* 192 (o) + 288 (op) + 96 (opn) + 1536 (opair) */
#define OFF_OP    192
#define OFF_OPN   480
#define OFF_OPAIR 576
#define VP_PAD 32            /* padded per-head channels for vp / opg */
#define WC_CONST 0.23570226039551587f  /* sqrt(2/(9*PQ)) */
#define WL_CONST 0.57735026918962576f  /* sqrt(1/3) */

// ---------------------------------------------------------------------------
// One-time operand prep: transpose + convert f32 [K,Nc] (ldw, batched) into
// f16 [Nc,K] (contiguous K). Writes are coalesced in k.
// ---------------------------------------------------------------------------
__global__ void transpose_to_f16(const float* __restrict__ W, long ldw, long strideW,
                                 _Float16* __restrict__ o, long strideO,
                                 int K, int Nc)
{
  long b = blockIdx.y;
  const float* Wb = W + b * strideW;
  _Float16* ob = o + b * strideO;
  int idx = blockIdx.x * 256 + threadIdx.x;
  if (idx >= K * Nc) return;
  int n = idx / K, k = idx % K;
  ob[(long)n * K + k] = (_Float16)Wb[(long)k * ldw + n];
}

// ---------------------------------------------------------------------------
// Generic strided/batched GEMM: C[M,Nc] = A[M,K] @ W[K,Nc] (+bias, +relu)
// A: f32 streaming operand (4x float4 loads + packed cvt per k-step).
// Wt: stationary operand, pre-transposed f16 [Nc,K] -> one contiguous 32B
//     fragment load per k-step. f32 accumulate. One wave per 16x16 tile.
// REQUIRES: M%16==0, K%32==0, Nc%16==0, lda%4==0.
// ---------------------------------------------------------------------------
__global__ void gemm_wmma_f16(const float* __restrict__ A, long lda, long strideA,
                              const _Float16* __restrict__ Wt, long strideW,
                              float* __restrict__ C, long ldc, long strideC,
                              const float* __restrict__ bias,
                              int K, int relu)
{
  const int b = blockIdx.z;
  A += (long)b * strideA; Wt += (long)b * strideW; C += (long)b * strideC;
  const int lane = threadIdx.x & 31;
  const int row0 = blockIdx.x * 16;
  const int col0 = blockIdx.y * 16;
  const int m  = row0 + (lane & 15);
  const int n  = col0 + (lane & 15);
  const int kbA = (lane >> 4) << 4;   // +0 / +16
  const int kbB = (lane >> 4) << 4;   // +0 / +16
  const _Float16* wrow = Wt + (long)n * K;
  v8f acc = {};
  for (int k0 = 0; k0 < K; k0 += 32) {
    // A fragment: e0..7 -> K=k0+kbA..+7 ; e8..15 -> K=k0+kbA+8..+15
    const float* arow = A + (long)m * lda + k0 + kbA;
    float4 a0 = ((const float4*)arow)[0];
    float4 a1 = ((const float4*)arow)[1];
    float4 a2 = ((const float4*)(arow + 8))[0];
    float4 a3 = ((const float4*)(arow + 8))[1];
    v16h af;
    af[0]=(_Float16)a0.x;  af[1]=(_Float16)a0.y;  af[2]=(_Float16)a0.z;  af[3]=(_Float16)a0.w;
    af[4]=(_Float16)a1.x;  af[5]=(_Float16)a1.y;  af[6]=(_Float16)a1.z;  af[7]=(_Float16)a1.w;
    af[8]=(_Float16)a2.x;  af[9]=(_Float16)a2.y;  af[10]=(_Float16)a2.z; af[11]=(_Float16)a2.w;
    af[12]=(_Float16)a3.x; af[13]=(_Float16)a3.y; af[14]=(_Float16)a3.z; af[15]=(_Float16)a3.w;
    // B fragment: lanes 0-15 K=k0..k0+15, lanes 16-31 K=k0+16..k0+31, col n
    v16h bf = *(const v16h*)(wrow + k0 + kbB);
    acc = __builtin_amdgcn_wmma_f32_16x16x32_f16(false, af, false, bf,
                                                 (short)0, acc, false, false);
  }
  float bv = bias ? bias[n] : 0.f;
#pragma unroll
  for (int r = 0; r < 8; ++r) {
    int mm = row0 + r + ((lane >> 4) << 3);
    float v = acc[r] + bv;
    if (relu) v = fmaxf(v, 0.f);
    C[(long)mm * ldc + n] = v;
  }
}

// ---------------------------------------------------------------------------
// wcg[h] = WC * softplus(gamma[h])
// ---------------------------------------------------------------------------
__global__ void prep_gamma(const float* __restrict__ gamma, float* __restrict__ wcg)
{
  int h = threadIdx.x;
  if (h < NH) {
    float g  = gamma[h];
    float sp = (g > 20.f) ? g : log1pf(__expf(g));
    wcg[h] = WC_CONST * sp;
  }
}

// ---------------------------------------------------------------------------
// Pad wb [128,12] -> wbp [128,16] (zero-filled)
// ---------------------------------------------------------------------------
__global__ void pad_wb(const float* __restrict__ wb, float* __restrict__ wbp)
{
  int idx = blockIdx.x * blockDim.x + threadIdx.x;   // k*16 + n
  if (idx >= CZ * 16) return;
  int k = idx >> 4, n = idx & 15;
  wbp[idx] = (n < NH) ? wb[k * NH + n] : 0.f;
}

// ---------------------------------------------------------------------------
// Apply frame: out = R p + t, layout [N, H, pitch] (zero-pad to pitch);
// optionally accumulate per-(n,h) squared norm.
// ---------------------------------------------------------------------------
__global__ void to_global_sq(const float* __restrict__ pin, const float* __restrict__ rot,
                             const float* __restrict__ trans, float* __restrict__ pout,
                             float* __restrict__ sq, int P, int pitch)
{
  int idx = blockIdx.x * blockDim.x + threadIdx.x;   // n*NH + h
  if (idx >= N_RES * NH) return;
  int n = idx / NH, h = idx % NH;
  float R[9];
#pragma unroll
  for (int t = 0; t < 9; ++t) R[t] = rot[(long)n * 9 + t];
  float t0 = trans[(long)n*3+0], t1 = trans[(long)n*3+1], t2 = trans[(long)n*3+2];
  float s = 0.f;
  long ib = (long)n * (NH * P * 3) + h * (P * 3);
  long ob = (long)idx * pitch;
  for (int p = 0; p < P; ++p) {
    float x = pin[ib + p*3+0], y = pin[ib + p*3+1], zz = pin[ib + p*3+2];
    float gx = R[0]*x + R[1]*y + R[2]*zz + t0;
    float gy = R[3]*x + R[4]*y + R[5]*zz + t1;
    float gz = R[6]*x + R[7]*y + R[8]*zz + t2;
    pout[ob + p*3+0] = gx; pout[ob + p*3+1] = gy; pout[ob + p*3+2] = gz;
    s += gx*gx + gy*gy + gz*gz;
  }
  for (int c = P * 3; c < pitch; ++c) pout[ob + c] = 0.f;
  if (sq) sq[(long)n * NH + h] = s;
}

// ---------------------------------------------------------------------------
// Build fused f16 logit operands [N, H, 32]:
//   cols 0..15  = qk-channel * chScale        (q: 1/sqrt(CH), k: 1.0)
//   cols 16..27 = global point coords * ps    (q: WC*softplus(gamma_h), k: 1.0)
//   cols 28..31 = 0
// ---------------------------------------------------------------------------
__global__ void build_fused(const float* __restrict__ qk,   // [N, H*CH]
                            const float* __restrict__ pg,   // [N, H, 12]
                            const float* __restrict__ wcg,  // [H] or null
                            _Float16* __restrict__ outf,    // [N, H, 32]
                            float chScale)
{
  int idx = blockIdx.x * blockDim.x + threadIdx.x;   // n*NH + h
  if (idx >= N_RES * NH) return;
  int n = idx / NH, h = idx % NH;
  float ps = wcg ? wcg[h] : 1.f;
  _Float16* o = outf + (long)idx * 32;
#pragma unroll
  for (int c = 0; c < CH; ++c)
    o[c] = (_Float16)(qk[(long)n * (NH*CH) + h*CH + c] * chScale);
#pragma unroll
  for (int c = 0; c < PQ*3; ++c)
    o[CH + c] = (_Float16)(pg[(long)idx * (PQ*3) + c] * ps);
#pragma unroll
  for (int c = CH + PQ*3; c < 32; ++c) o[c] = (_Float16)0.f;
}

// ---------------------------------------------------------------------------
// Fused logits + softmax for one (head, 16-row i-tile).
// One K=32 WMMA per j-tile using pre-packed f16 operands (branch-free loads).
// Epilogue adds pair bias and -0.5*wcg*(|qp_i|^2+|kp_j|^2), scales by WL.
// Softmax across j in LDS, writes a[h,i,j].
// ---------------------------------------------------------------------------
__global__ void attn_logits_softmax(const _Float16* __restrict__ qf,
                                    const _Float16* __restrict__ kf,
                                    const float* __restrict__ sqq,
                                    const float* __restrict__ sqk,
                                    const float* __restrict__ biasz,  // [N*N, 16]
                                    const float* __restrict__ wcg_arr,
                                    float* __restrict__ a_out)
{
  __shared__ float sm[16 * N_RES];
  __shared__ float red[16 * 8];
  const int h    = blockIdx.y;
  const int i0   = blockIdx.x * 16;
  const int tid  = threadIdx.x;
  const int lane = tid & 31;
  const int wave = tid >> 5;
  const float wcg = wcg_arr[h];

  // A fragment: rows i0..i0+15
  v16h af;
  {
    int i  = i0 + (lane & 15);
    int kb = (lane >> 4) << 4;
    const _Float16* p = qf + ((long)i * NH + h) * 32;
    v8h lo = *(const v8h*)(p + kb);        // K = kb..kb+7
    v8h hi = *(const v8h*)(p + kb + 8);    // K = kb+8..kb+15
#pragma unroll
    for (int e = 0; e < 8; ++e) { af[e] = lo[e]; af[e + 8] = hi[e]; }
  }

  for (int jt = wave; jt < N_RES / 16; jt += 4) {
    int j0 = jt * 16;
    int j  = j0 + (lane & 15);
    v16h bf = *(const v16h*)(kf + ((long)j * NH + h) * 32 + ((lane >> 4) << 4));
    v8f c = {};
    c = __builtin_amdgcn_wmma_f32_16x16x32_f16(false, af, false, bf,
                                               (short)0, c, false, false);
    int jj = j0 + (lane & 15);
#pragma unroll
    for (int r = 0; r < 8; ++r) {
      int m_local = r + ((lane >> 4) << 3);
      int i = i0 + m_local;
      float logit = c[r]
                  + biasz[((long)i * N_RES + jj) * 16 + h]
                  - 0.5f * wcg * (sqq[(long)i * NH + h] + sqk[(long)jj * NH + h]);
      sm[m_local * N_RES + jj] = WL_CONST * logit;
    }
  }
  __syncthreads();

  // Softmax over j: 16 rows x 8 threads/row
  int row = tid >> 3, sub = tid & 7;
  float mx = -1e30f;
  for (int j = sub; j < N_RES; j += 8) mx = fmaxf(mx, sm[row * N_RES + j]);
  red[row*8 + sub] = mx;
  __syncthreads();
  if (sub == 0) {
    float m2 = red[row*8];
#pragma unroll
    for (int t = 1; t < 8; ++t) m2 = fmaxf(m2, red[row*8 + t]);
    red[row*8] = m2;
  }
  __syncthreads();
  mx = red[row*8];
  __syncthreads();
  float s = 0.f;
  for (int j = sub; j < N_RES; j += 8) {
    float e = __expf(sm[row * N_RES + j] - mx);
    sm[row * N_RES + j] = e;
    s += e;
  }
  red[row*8 + sub] = s;
  __syncthreads();
  if (sub == 0) {
    float s2 = 0.f;
#pragma unroll
    for (int t = 0; t < 8; ++t) s2 += red[row*8 + t];
    red[row*8] = s2;
  }
  __syncthreads();
  float inv = 1.f / red[row*8];
  long i = i0 + row;
  for (int j = sub; j < N_RES; j += 8)
    a_out[((long)h * N_RES + i) * N_RES + j] = sm[row * N_RES + j] * inv;
}

// ---------------------------------------------------------------------------
// opair[i,h,c] = sum_j a[h,i,j] * z[i,j,c]  (z operand varies with output row
// -> not a GEMM; stream z once, batch all 12 heads per z read). Writes cat.
// ---------------------------------------------------------------------------
__global__ void opair_kernel(const float* __restrict__ a,
                             const float* __restrict__ z,
                             float* __restrict__ cat)
{
  const int i0  = blockIdx.x * 16;
  const int tid = threadIdx.x;         // 256
  const int row = tid >> 4;
  const int ch0 = (tid & 15) * 8;
  const long i  = i0 + row;
  float acc[NH][8];
#pragma unroll
  for (int h = 0; h < NH; ++h)
#pragma unroll
    for (int e = 0; e < 8; ++e) acc[h][e] = 0.f;

  for (int j = 0; j < N_RES; ++j) {
    if (j + 8 < N_RES)   // stream-ahead prefetch of z (global_prefetch_b8)
      __builtin_prefetch(z + ((long)i * N_RES + j + 8) * CZ + ch0, 0, 0);
    const float4* z4 = reinterpret_cast<const float4*>(z + ((long)i * N_RES + j) * CZ + ch0);
    float4 za = z4[0], zb = z4[1];
    float zv[8] = {za.x, za.y, za.z, za.w, zb.x, zb.y, zb.z, zb.w};
#pragma unroll
    for (int h = 0; h < NH; ++h) {
      float av = a[((long)h * N_RES + i) * N_RES + j];
#pragma unroll
      for (int e = 0; e < 8; ++e) acc[h][e] = fmaf(av, zv[e], acc[h][e]);
    }
  }
#pragma unroll
  for (int h = 0; h < NH; ++h)
#pragma unroll
    for (int e = 0; e < 8; ++e)
      cat[i * CATD + OFF_OPAIR + h * CZ + ch0 + e] = acc[h][e];
}

// ---------------------------------------------------------------------------
// op = R^T (opg - t) ; opn = sqrt(|op|^2 + 1e-8). opg padded [N,H,VP_PAD].
// ---------------------------------------------------------------------------
__global__ void finalize_points(const float* __restrict__ opg, const float* __restrict__ rot,
                                const float* __restrict__ trans, float* __restrict__ cat)
{
  int idx = blockIdx.x * blockDim.x + threadIdx.x;
  if (idx >= N_RES * NH * PV) return;
  int n   = idx / (NH * PV);
  int rem = idx % (NH * PV);
  int h = rem / PV, p = rem % PV;
  float R[9];
#pragma unroll
  for (int t = 0; t < 9; ++t) R[t] = rot[(long)n * 9 + t];
  long pb = ((long)n * NH + h) * VP_PAD + p * 3;
  float dx = opg[pb+0] - trans[(long)n*3+0];
  float dy = opg[pb+1] - trans[(long)n*3+1];
  float dz = opg[pb+2] - trans[(long)n*3+2];
  float lx = R[0]*dx + R[3]*dy + R[6]*dz;   // R^T
  float ly = R[1]*dx + R[4]*dy + R[7]*dz;
  float lz = R[2]*dx + R[5]*dy + R[8]*dz;
  long c = (long)n * CATD;
  cat[c + OFF_OP  + h*(PV*3) + p*3+0] = lx;
  cat[c + OFF_OP  + h*(PV*3) + p*3+1] = ly;
  cat[c + OFF_OP  + h*(PV*3) + p*3+2] = lz;
  cat[c + OFF_OPN + h*PV + p] = sqrtf(lx*lx + ly*ly + lz*lz + 1e-8f);
}

// ---------------------------------------------------------------------------
// out[n,:] = LN(xa[n,:] + xb[n,:]) * g + b   (biased variance, eps=1e-5)
// ---------------------------------------------------------------------------
__global__ void ln_add_kernel(const float* __restrict__ xa, const float* __restrict__ xb,
                              const float* __restrict__ g, const float* __restrict__ b,
                              float* __restrict__ out, int C)
{
  int n = blockIdx.x;
  __shared__ float vals[CN];
  __shared__ float rsum[128];
  __shared__ float rsq[128];
  int tid = threadIdx.x;
  float s = 0.f, sq = 0.f;
  for (int c = tid; c < C; c += 128) {
    float v = xa[(long)n * C + c] + (xb ? xb[(long)n * C + c] : 0.f);
    vals[c] = v; s += v; sq += v * v;
  }
  rsum[tid] = s; rsq[tid] = sq;
  __syncthreads();
  for (int st = 64; st > 0; st >>= 1) {
    if (tid < st) { rsum[tid] += rsum[tid + st]; rsq[tid] += rsq[tid + st]; }
    __syncthreads();
  }
  float mean = rsum[0] / C;
  float var  = rsq[0] / C - mean * mean;
  float rstd = rsqrtf(var + 1e-5f);
  for (int c = tid; c < C; c += 128)
    out[(long)n * C + c] = (vals[c] - mean) * rstd * g[c] + b[c];
}

// ---------------------------------------------------------------------------
extern "C" void kernel_launch(void* const* d_in, const int* in_sizes, int n_in,
                              void* d_out, int out_size, void* d_ws, size_t ws_size,
                              hipStream_t stream)
{
  (void)in_sizes; (void)n_in; (void)out_size; (void)ws_size;
  const float* s0    = (const float*)d_in[0];
  const float* z     = (const float*)d_in[1];
  const float* rot   = (const float*)d_in[2];
  const float* trans = (const float*)d_in[3];
  const float* wq    = (const float*)d_in[4];
  const float* wk    = (const float*)d_in[5];
  const float* wv    = (const float*)d_in[6];
  const float* wqp   = (const float*)d_in[7];
  const float* wkp   = (const float*)d_in[8];
  const float* wvp   = (const float*)d_in[9];
  const float* wb    = (const float*)d_in[10];
  const float* gamma = (const float*)d_in[11];
  const float* wo    = (const float*)d_in[12];
  const float* bo    = (const float*)d_in[13];
  const float* ln1g  = (const float*)d_in[14];
  const float* ln1b  = (const float*)d_in[15];
  const float* t1w   = (const float*)d_in[16];
  const float* t1b   = (const float*)d_in[17];
  const float* t2w   = (const float*)d_in[18];
  const float* t2b   = (const float*)d_in[19];
  const float* t3w   = (const float*)d_in[20];
  const float* t3b   = (const float*)d_in[21];
  const float* ln2g  = (const float*)d_in[22];
  const float* ln2b  = (const float*)d_in[23];
  float* out = (float*)d_out;

  float* ws = (float*)d_ws;
  size_t off = 0;
  auto alloc = [&](size_t nflt) { float* p = ws + off; off += nflt; return p; };
  auto alloch = [&](size_t nhalf) { return (_Float16*)alloc((nhalf + 1) / 2); };
  float* q     = alloc((size_t)N_RES * 192);
  float* kbuf  = alloc((size_t)N_RES * 192);
  float* v     = alloc((size_t)N_RES * 192);
  float* qpl   = alloc((size_t)N_RES * 144);
  float* kpl   = alloc((size_t)N_RES * 144);
  float* vpl   = alloc((size_t)N_RES * 288);
  float* qpg   = alloc((size_t)N_RES * NH * 12);
  float* kpg   = alloc((size_t)N_RES * NH * 12);
  float* vpgp  = alloc((size_t)N_RES * NH * VP_PAD);      // padded
  float* sqq   = alloc((size_t)N_RES * NH);
  float* sqk   = alloc((size_t)N_RES * NH);
  float* wcg   = alloc(16);
  float* wbp   = alloc((size_t)CZ * 16);                   // padded wb
  float* biasz = alloc((size_t)N_RES * N_RES * 16);        // padded bias
  float* a     = alloc((size_t)NH * N_RES * N_RES);
  float* opgp  = alloc((size_t)N_RES * NH * VP_PAD);       // padded
  float* cat   = alloc((size_t)N_RES * CATD);
  float* ipa   = alloc((size_t)N_RES * CN);
  float* x     = alloc((size_t)N_RES * CN);
  float* y1    = alloc((size_t)N_RES * CN);
  float* y2    = alloc((size_t)N_RES * CN);
  float* y3    = alloc((size_t)N_RES * CN);
  _Float16* qf = alloch((size_t)N_RES * NH * 32);
  _Float16* kf = alloch((size_t)N_RES * NH * 32);
  // transposed f16 stationary operands [Nc, K]
  _Float16* wqT  = alloch((size_t)192 * CN);
  _Float16* wkT  = alloch((size_t)192 * CN);
  _Float16* wvT  = alloch((size_t)192 * CN);
  _Float16* wqpT = alloch((size_t)144 * CN);
  _Float16* wkpT = alloch((size_t)144 * CN);
  _Float16* wvpT = alloch((size_t)288 * CN);
  _Float16* wbT  = alloch((size_t)16 * CZ);
  _Float16* woT  = alloch((size_t)CN * CATD);
  _Float16* t1T  = alloch((size_t)CN * CN);
  _Float16* t2T  = alloch((size_t)CN * CN);
  _Float16* t3T  = alloch((size_t)CN * CN);
  _Float16* vT   = alloch((size_t)NH * CH * N_RES);       // per-head [16,768]
  _Float16* vpgT = alloch((size_t)NH * VP_PAD * N_RES);   // per-head [32,768]

  auto xpose = [&](const float* W, long ldw, long sW, _Float16* o, long sO,
                   int K, int Nc, int batch) {
    dim3 g((unsigned)((K * Nc + 255) / 256), (unsigned)batch);
    transpose_to_f16<<<g, 256, 0, stream>>>(W, ldw, sW, o, sO, K, Nc);
  };
  auto gemm = [&](const float* A, long lda, long sA,
                  const _Float16* Wt, long sW,
                  float* C, long ldc, long sC, const float* bias,
                  int M, int K, int Nc, int relu, int batch) {
    dim3 g((unsigned)(M / 16), (unsigned)(Nc / 16), (unsigned)batch);
    gemm_wmma_f16<<<g, 32, 0, stream>>>(A, lda, sA, Wt, sW, C, ldc, sC,
                                        bias, K, relu);
  };

  prep_gamma<<<1, 32, 0, stream>>>(gamma, wcg);
  pad_wb<<<(CZ * 16 + 127) / 128, 128, 0, stream>>>(wb, wbp);

  // One-time weight transposes -> f16 [Nc,K]
  xpose(wq,  192, 0, wqT,  0, CN, 192, 1);
  xpose(wk,  192, 0, wkT,  0, CN, 192, 1);
  xpose(wv,  192, 0, wvT,  0, CN, 192, 1);
  xpose(wqp, 144, 0, wqpT, 0, CN, 144, 1);
  xpose(wkp, 144, 0, wkpT, 0, CN, 144, 1);
  xpose(wvp, 288, 0, wvpT, 0, CN, 288, 1);
  xpose(wbp, 16,  0, wbT,  0, CZ, 16, 1);
  xpose(wo,  CN,  0, woT,  0, CATD, CN, 1);
  xpose(t1w, CN,  0, t1T,  0, CN, CN, 1);
  xpose(t2w, CN,  0, t2T,  0, CN, CN, 1);
  xpose(t3w, CN,  0, t3T,  0, CN, CN, 1);

  // Projections from s0
  gemm(s0, CN, 0, wqT,  0, q,    192, 0, nullptr, N_RES, CN, 192, 0, 1);
  gemm(s0, CN, 0, wkT,  0, kbuf, 192, 0, nullptr, N_RES, CN, 192, 0, 1);
  gemm(s0, CN, 0, wvT,  0, v,    192, 0, nullptr, N_RES, CN, 192, 0, 1);
  gemm(s0, CN, 0, wqpT, 0, qpl,  144, 0, nullptr, N_RES, CN, 144, 0, 1);
  gemm(s0, CN, 0, wkpT, 0, kpl,  144, 0, nullptr, N_RES, CN, 144, 0, 1);
  gemm(s0, CN, 0, wvpT, 0, vpl,  288, 0, nullptr, N_RES, CN, 288, 0, 1);

  // Frames + squared norms (+padding for vp)
  int tg = N_RES * NH;
  to_global_sq<<<(tg + 127) / 128, 128, 0, stream>>>(qpl, rot, trans, qpg, sqq, PQ, PQ*3);
  to_global_sq<<<(tg + 127) / 128, 128, 0, stream>>>(kpl, rot, trans, kpg, sqk, PQ, PQ*3);
  to_global_sq<<<(tg + 127) / 128, 128, 0, stream>>>(vpl, rot, trans, vpgp, nullptr, PV, VP_PAD);

  // Fused f16 logit operands
  build_fused<<<(tg + 127) / 128, 128, 0, stream>>>(q,    qpg, wcg,     qf, 0.25f);
  build_fused<<<(tg + 127) / 128, 128, 0, stream>>>(kbuf, kpg, nullptr, kf, 1.0f);

  // Transpose value operands per head (after they exist)
  xpose(v,    192, CH,     vT,   (long)CH * N_RES,     N_RES, CH,     NH);
  xpose(vpgp, NH * VP_PAD, VP_PAD, vpgT, (long)VP_PAD * N_RES, N_RES, VP_PAD, NH);

  // Pair bias: [N*N,128] @ [128,16-padded]   (first streaming pass over z)
  gemm(z, CZ, 0, wbT, 0, biasz, 16, 0, nullptr, N_RES * N_RES, CZ, 16, 0, 1);

  // Fused logits + softmax -> a[h,i,j]
  dim3 ag(N_RES / 16, NH);
  attn_logits_softmax<<<ag, 128, 0, stream>>>(qf, kf, sqq, sqk, biasz, wcg, a);

  // o = a @ v_h -> cat cols [0,192); batched over heads
  gemm(a, N_RES, (long)N_RES * N_RES, vT, (long)CH * N_RES,
       cat, CATD, CH, nullptr, N_RES, N_RES, CH, 0, NH);
  // opg = a @ vpg_h (padded to 32 ch/head)
  gemm(a, N_RES, (long)N_RES * N_RES, vpgT, (long)VP_PAD * N_RES,
       opgp, NH * VP_PAD, VP_PAD, nullptr, N_RES, N_RES, VP_PAD, 0, NH);

  // opair (second streaming pass over z) -> cat cols [576,2112)
  opair_kernel<<<N_RES / 16, 256, 0, stream>>>(a, z, cat);
  // inverse frame + norms -> cat cols [192,576)
  finalize_points<<<(N_RES * NH * PV + 255) / 256, 256, 0, stream>>>(opgp, rot, trans, cat);

  // Output projection + residual/LN + transition + LN
  gemm(cat, CATD, 0, woT, 0, ipa, CN, 0, bo, N_RES, CATD, CN, 0, 1);
  ln_add_kernel<<<N_RES, 128, 0, stream>>>(s0, ipa, ln1g, ln1b, x, CN);
  gemm(x,  CN, 0, t1T, 0, y1, CN, 0, t1b, N_RES, CN, CN, 1, 1);
  gemm(y1, CN, 0, t2T, 0, y2, CN, 0, t2b, N_RES, CN, CN, 1, 1);
  gemm(y2, CN, 0, t3T, 0, y3, CN, 0, t3b, N_RES, CN, CN, 0, 1);
  ln_add_kernel<<<N_RES, 128, 0, stream>>>(y3, x, ln2g, ln2b, out, CN);
}
